// EPMoE_5978594476597
// MI455X (gfx1250) — matrix operand order repaired and checked by reading, not compiled, not run on hardware
//
#include <hip/hip_runtime.h>
#include <hip/hip_bf16.h>

// ---------------- problem constants (match reference) ----------------
#define T_TOK   2048
#define HIDDEN  2048
#define INTER   1408
#define NEXP    64
#define TOPK    6
#define EPSZ    8
#define NLOCAL  8            // NUM_EXPERTS / EP_SIZE
#define SC13_R  22           // 2*INTER/128
#define SC13_C  16           // HIDDEN/128
#define SC2_R   16           // HIDDEN/128
#define SC2_C   11           // INTER/128
#define KPANEL  128          // K staged per barrier pair (== scale block K)
#define LDSPAD  136          // row stride in bf16 (272B = 68 banks -> conflict-free)
#define MTILE   64           // tokens per block (4 x 16-row WMMA sub-tiles)

typedef __attribute__((ext_vector_type(16))) __bf16 v16bf;
typedef __attribute__((ext_vector_type(8)))  __bf16 v8bf;
typedef __attribute__((ext_vector_type(4)))  __bf16 v4bf;
typedef __attribute__((ext_vector_type(8)))  float  v8f;

// ---------------- workspace layout ----------------
// [0,32)          : int cnt[8]
// [64, +64K)      : int tok[8][2048]
// [+64K, +128K)   : float wt[8][2048]
// [+128K, ~46MB)  : __bf16 act[8][2048][1408]
#define WS_TOK_OFF   64
#define WS_WT_OFF    (64 + 8 * T_TOK * 4)
#define WS_ACT_OFF   (64 + 2 * 8 * T_TOK * 4)

// =====================================================================
__global__ void moe_zero(float* __restrict__ out, int n, int* __restrict__ cnt) {
    int i = blockIdx.x * blockDim.x + threadIdx.x;
    if (i < n) out[i] = 0.0f;
    if (i < NLOCAL) cnt[i] = 0;
}

// =====================================================================
// Router: one thread per token; top-6 insertion scan. Softmax denominator
// cancels under top-k renormalization: w_i = exp(l_i-m)/sum_topk exp(l_j-m).
// =====================================================================
__global__ __launch_bounds__(256)
void moe_router(const float* __restrict__ logits, int* __restrict__ cnt,
                int* __restrict__ tok, float* __restrict__ wt) {
    int t = blockIdx.x * blockDim.x + threadIdx.x;
    if (t >= T_TOK) return;
    const float* row = logits + (size_t)t * NEXP;

    float bv[TOPK]; int bi[TOPK];
#pragma unroll
    for (int i = 0; i < TOPK; ++i) { bv[i] = -3.4e38f; bi[i] = -1; }
    for (int e = 0; e < NEXP; ++e) {
        float v = row[e];
        if (v > bv[TOPK - 1]) {
            int p = TOPK - 1;
            while (p > 0 && bv[p - 1] < v) { bv[p] = bv[p - 1]; bi[p] = bi[p - 1]; --p; }
            bv[p] = v; bi[p] = e;
        }
    }
    float m = bv[0], denom = 0.0f;
#pragma unroll
    for (int i = 0; i < TOPK; ++i) denom += __expf(bv[i] - m);
    float inv = 1.0f / denom;
#pragma unroll
    for (int i = 0; i < TOPK; ++i) {
        int id = bi[i];
        if ((id % EPSZ) == 0) {                 // EP_RANK == 0
            int lid = id / EPSZ;
            int pos = atomicAdd(&cnt[lid], 1);
            tok[lid * T_TOK + pos] = t;
            wt [lid * T_TOK + pos] = __expf(bv[i] - m) * inv;
        }
    }
}

// ---------------------------------------------------------------------
// B fragment: 16 consecutive f32 weights, dequant (block scale) folded into
// the f32->bf16 conversion. A fragment: two 8-bf16 runs from LDS.
// A and B both use contiguous k = 16*(lane/16) + s (lanes 0-15: K 0..15, lanes 16-31: K 16..31).
// ---------------------------------------------------------------------
__device__ __forceinline__ v16bf load_bfrag(const float* bp, float s) {
    const float4* bp4 = (const float4*)bp;
    float4 b0 = bp4[0], b1 = bp4[1], b2 = bp4[2], b3 = bp4[3];
    float f[16] = { b0.x, b0.y, b0.z, b0.w, b1.x, b1.y, b1.z, b1.w,
                    b2.x, b2.y, b2.z, b2.w, b3.x, b3.y, b3.z, b3.w };
    v16bf r;
#pragma unroll
    for (int j = 0; j < 16; ++j) r[j] = (__bf16)(f[j] * s);
    return r;
}

__device__ __forceinline__ v16bf load_afrag(const __bf16* rowp, int kk, int kbase) {
    v8bf lo = *(const v8bf*)(rowp + kk + kbase);
    v8bf hi = *(const v8bf*)(rowp + kk + kbase + 8);
    v16bf r;
#pragma unroll
    for (int j = 0; j < 8; ++j) { r[j] = lo[j]; r[j + 8] = hi[j]; }
    return r;
}

// =====================================================================
// GEMM1 + fused silu(gate)*up. Block = 8 waves over a 64-token M-tile.
// Waves 0-3: 64 gate cols; waves 4-7: matching 64 up cols.
// K staged in 128-wide panels; per panel: 4 k-steps x 4 row sub-tiles
// = 16 WMMAs; each B fragment is reused by 4 independent accumulators.
// =====================================================================
__global__ __launch_bounds__(256)
void moe_gemm1(const float* __restrict__ x, const float* __restrict__ w13,
               const float* __restrict__ w13s, const int* __restrict__ cnt,
               const int* __restrict__ tok, __bf16* __restrict__ act) {
    const int e  = blockIdx.z;
    const int mb = blockIdx.y;
    const int nb = blockIdx.x;                 // 22 blocks of 64 gate cols
    const int count = cnt[e];
    if (mb * MTILE >= count) return;           // block-uniform: EXEC stays full

    __shared__ __align__(16) __bf16 sA[MTILE][LDSPAD];   // 64 x 128 K (17 KB)
    __shared__ float sUp[4][4][32][8];                   // up accs (16 KB)
    __shared__ int   stok[MTILE];

    const int tid  = threadIdx.x;
    const int wave = tid >> 5;
    const int lane = tid & 31;
    const bool isUp = wave >= 4;

    if (tid < MTILE) {
        int slot = mb * MTILE + tid;
        stok[tid] = (slot < count) ? tok[e * T_TOK + slot] : tok[e * T_TOK];
    }

    const int n = nb * 64 + (wave & 3) * 16 + (lane & 15) + (isUp ? INTER : 0);
    const float* brow = w13 + ((size_t)e * 2 * INTER + n) * (size_t)HIDDEN;
    const int sn    = n >> 7;
    const int khalf = (lane >> 4) * 16;
    const int M     = lane & 15;
    const int kbase = (lane < 16) ? 0 : 16;

    v8f acc[4] = {};
    for (int k0 = 0; k0 < HIDDEN; k0 += KPANEL) {        // 16 panels
        __syncthreads();
        // stage A panel: 2048 float4 loads -> bf16 -> LDS (8 per thread)
#pragma unroll
        for (int i = tid; i < MTILE * 32; i += 256) {
            int r = i >> 5, c4 = i & 31;
            float4 v = *(const float4*)(x + (size_t)stok[r] * HIDDEN + k0 + c4 * 4);
            v4bf h; h[0] = (__bf16)v.x; h[1] = (__bf16)v.y;
                    h[2] = (__bf16)v.z; h[3] = (__bf16)v.w;
            *(v4bf*)&sA[r][c4 * 4] = h;
        }
        __syncthreads();

        const float s = w13s[(e * SC13_R + sn) * SC13_C + (k0 >> 7)];
        __builtin_prefetch((const void*)(brow + k0 + KPANEL), 0, 1);
#pragma unroll
        for (int kk = 0; kk < KPANEL; kk += 32) {
            v16bf bfrag = load_bfrag(brow + k0 + kk + khalf, s);
#pragma unroll
            for (int st = 0; st < 4; ++st) {             // 4 independent accs
                v16bf afrag = load_afrag(&sA[st * 16 + M][0], kk, kbase);
                acc[st] = __builtin_amdgcn_wmma_f32_16x16x32_bf16(
                              false, afrag, false, bfrag, (short)0, acc[st],
                              false, false);
            }
        }
    }

    if (isUp) {
#pragma unroll
        for (int st = 0; st < 4; ++st)
#pragma unroll
            for (int r = 0; r < 8; ++r) sUp[wave - 4][st][lane][r] = acc[st][r];
    }
    __syncthreads();
    if (!isUp) {
        const int M0   = (lane < 16) ? 0 : 8;
        const int colN = nb * 64 + wave * 16 + (lane & 15);
#pragma unroll
        for (int st = 0; st < 4; ++st) {
#pragma unroll
            for (int r = 0; r < 8; ++r) {
                float g = acc[st][r];
                float u = sUp[wave][st][lane][r];
                float a = (g / (1.0f + __expf(-g))) * u;   // silu(g) * u
                int row = mb * MTILE + st * 16 + M0 + r;
                act[((size_t)e * T_TOK + row) * INTER + colN] = (__bf16)a;
            }
        }
    }
}

// =====================================================================
// GEMM2: act @ w2^T, scaled by combine weight, f32 global atomics into out.
// Same 64-row M-tile / 4-accumulator structure.
// =====================================================================
__global__ __launch_bounds__(256)
void moe_gemm2(const __bf16* __restrict__ act, const float* __restrict__ w2,
               const float* __restrict__ w2s, const int* __restrict__ cnt,
               const int* __restrict__ tok, const float* __restrict__ wt,
               float* __restrict__ out) {
    const int e  = blockIdx.z;
    const int mb = blockIdx.y;
    const int nb = blockIdx.x;                 // 16 blocks of 128 cols
    const int count = cnt[e];
    if (mb * MTILE >= count) return;

    __shared__ __align__(16) __bf16 sA[MTILE][LDSPAD];

    const int tid  = threadIdx.x;
    const int wave = tid >> 5;
    const int lane = tid & 31;

    const int n = nb * 128 + wave * 16 + (lane & 15);
    const float* brow = w2 + ((size_t)e * HIDDEN + n) * (size_t)INTER;
    const int sn    = n >> 7;
    const int khalf = (lane >> 4) * 16;
    const int M     = lane & 15;
    const int kbase = (lane < 16) ? 0 : 16;

    v8f acc[4] = {};
    for (int k0 = 0; k0 < INTER; k0 += KPANEL) {         // 11 panels
        __syncthreads();
        // stage A panel: straight bf16 copy, 64 rows x 16 uint4 (4 per thread)
#pragma unroll
        for (int i = tid; i < MTILE * 16; i += 256) {
            int r = i >> 4, c = i & 15;
            uint4 v = *(const uint4*)(act + ((size_t)e * T_TOK + mb * MTILE + r) * INTER
                                          + k0 + c * 8);
            *(uint4*)&sA[r][c * 8] = v;
        }
        __syncthreads();

        const float s = w2s[(e * SC2_R + sn) * SC2_C + (k0 >> 7)];
        __builtin_prefetch((const void*)(brow + k0 + KPANEL), 0, 1);
#pragma unroll
        for (int kk = 0; kk < KPANEL; kk += 32) {
            v16bf bfrag = load_bfrag(brow + k0 + kk + khalf, s);
#pragma unroll
            for (int st = 0; st < 4; ++st) {
                v16bf afrag = load_afrag(&sA[st * 16 + M][0], kk, kbase);
                acc[st] = __builtin_amdgcn_wmma_f32_16x16x32_bf16(
                              false, afrag, false, bfrag, (short)0, acc[st],
                              false, false);
            }
        }
    }

    const int M0 = (lane < 16) ? 0 : 8;
#pragma unroll
    for (int st = 0; st < 4; ++st) {
#pragma unroll
        for (int r = 0; r < 8; ++r) {
            int slot = mb * MTILE + st * 16 + M0 + r;
            if (slot < count) {
                int   t = tok[e * T_TOK + slot];
                float w = wt [e * T_TOK + slot];
                __hip_atomic_fetch_add(&out[(size_t)t * HIDDEN + n], w * acc[st][r],
                                       __ATOMIC_RELAXED, __HIP_MEMORY_SCOPE_AGENT);
            }
        }
    }
}

// =====================================================================
extern "C" void kernel_launch(void* const* d_in, const int* in_sizes, int n_in,
                              void* d_out, int out_size, void* d_ws, size_t ws_size,
                              hipStream_t stream) {
    const float* x      = (const float*)d_in[0];
    const float* logits = (const float*)d_in[1];
    const float* w13    = (const float*)d_in[2];
    const float* w13s   = (const float*)d_in[3];
    const float* w2     = (const float*)d_in[4];
    const float* w2s    = (const float*)d_in[5];
    float* out = (float*)d_out;

    char* ws = (char*)d_ws;
    int*    cnt = (int*)ws;
    int*    tok = (int*)(ws + WS_TOK_OFF);
    float*  wtl = (float*)(ws + WS_WT_OFF);
    __bf16* act = (__bf16*)(ws + WS_ACT_OFF);   // needs ~46.3 MB of ws

    moe_zero<<<(out_size + 255) / 256, 256, 0, stream>>>(out, out_size, cnt);
    moe_router<<<(T_TOK + 255) / 256, 256, 0, stream>>>(logits, cnt, tok, wtl);
    moe_gemm1<<<dim3(INTER / 64, T_TOK / MTILE, NLOCAL), 256, 0, stream>>>(
        x, w13, w13s, cnt, tok, act);
    moe_gemm2<<<dim3(HIDDEN / 128, T_TOK / MTILE, NLOCAL), 256, 0, stream>>>(
        act, w2, w2s, cnt, tok, wtl, out);
}
